// TemporalResonanceLayer_27092653703889
// MI455X (gfx1250) — matrix operand order repaired and checked by reading, not compiled, not run on hardware
//
#include <hip/hip_runtime.h>
#include <hip/hip_bf16.h>

// ---------------------------------------------------------------------------
// TemporalResonanceLayer on MI455X (gfx1250, wave32, WMMA 16x16x32 f16)
//   B=4, n=4096, D=1024, K=256, M = B*n = 16384
// Pipeline:
//   0) pack W_proj|W_phase -> f16 Wcat[1792][1024]; W_res -> f16 Wres[1024][1024] (pad)
//   1) k_proj : x @ Wcat^T via v_wmma (7 fused accumulators: 6 chunks + phase),
//               epilogue computes alpha, drive, angle, beta, g, pq  (f32 ws)
//   2) k_scan_chunks / k_scan_prefix / k_apply : chunked parallel complex
//      recurrence r_t = a_t r_{t-1} + d_t, apply pass fused with
//      erase / norm / demodulate / phase-gate / SCPM -> rho f16 [16384][1024]
//   3) k_out  : rho @ Wres^T via v_wmma, scaled by res_scale -> d_out (f32)
// ---------------------------------------------------------------------------

typedef __attribute__((ext_vector_type(16))) _Float16 v16h;
typedef __attribute__((ext_vector_type(8)))  float    v8f;

#define DM   1024
#define KOSC 256
#define NSEQ 4096
#define NB   4
#define MROW 16384          // NB*NSEQ
#define NCHUNK 32           // scan chunks per batch
#define CHLEN  128          // NSEQ / NCHUNK

__device__ __forceinline__ float sigmoidf_(float x) { return 1.f / (1.f + __expf(-x)); }
__device__ __forceinline__ float softplusf_(float x) {
    return fmaxf(x, 0.f) + log1pf(__expf(-fabsf(x)));
}

// A fragment (16x32, f16 elems) from f32 row-major source; p = row base + kstep base
__device__ __forceinline__ v16h load_a_f32(const float* __restrict__ p, int g) {
    const float4* q0 = reinterpret_cast<const float4*>(p + 8 * g);
    const float4* q1 = reinterpret_cast<const float4*>(p + 16 + 8 * g);
    float4 f0 = q0[0], f1 = q0[1], f2 = q1[0], f3 = q1[1];
    v16h a;
    a[0]=(_Float16)f0.x; a[1]=(_Float16)f0.y; a[2]=(_Float16)f0.z; a[3]=(_Float16)f0.w;
    a[4]=(_Float16)f1.x; a[5]=(_Float16)f1.y; a[6]=(_Float16)f1.z; a[7]=(_Float16)f1.w;
    a[8]=(_Float16)f2.x; a[9]=(_Float16)f2.y; a[10]=(_Float16)f2.z; a[11]=(_Float16)f2.w;
    a[12]=(_Float16)f3.x; a[13]=(_Float16)f3.y; a[14]=(_Float16)f3.z; a[15]=(_Float16)f3.w;
    return a;
}

// A fragment from f16 source (two contiguous 8-half runs)
__device__ __forceinline__ v16h load_a_f16(const _Float16* __restrict__ p, int g) {
    union { uint4 u[2]; v16h h; } u;
    u.u[0] = *reinterpret_cast<const uint4*>(p + 8 * g);
    u.u[1] = *reinterpret_cast<const uint4*>(p + 16 + 8 * g);
    return u.h;
}

// B fragment (32x16): lane holds two 8-half K runs at p + 8*g and p + 16 + 8*g
__device__ __forceinline__ v16h load_b_f16(const _Float16* __restrict__ p, int g) {
    union { uint4 u[2]; v16h h; } u;
    u.u[0] = *reinterpret_cast<const uint4*>(p + 8 * g);
    u.u[1] = *reinterpret_cast<const uint4*>(p + 16 + 8 * g);
    return u.h;
}

// ---------------------------------------------------------------------------
// 0) weight pack kernels (f32 -> f16, concatenate / pad)
// ---------------------------------------------------------------------------
__global__ void k_pack_wcat(const float* __restrict__ Wp, const float* __restrict__ Wph,
                            _Float16* __restrict__ Wcat) {
    int i = blockIdx.x * blockDim.x + threadIdx.x;          // 1792*1024
    if (i >= 1792 * DM) return;
    int row = i >> 10, d = i & (DM - 1);
    float v = (row < 1536) ? Wp[i] : Wph[(size_t)(row - 1536) * DM + d];
    Wcat[i] = (_Float16)v;
}

__global__ void k_pack_wres(const float* __restrict__ Wr, _Float16* __restrict__ Wres) {
    int i = blockIdx.x * blockDim.x + threadIdx.x;          // 1024*1024
    if (i >= DM * DM) return;
    int d = i >> 10, kk = i & (DM - 1);
    Wres[i] = (kk < 1022) ? (_Float16)Wr[(size_t)d * 1022 + kk] : (_Float16)0.f;
}

// ---------------------------------------------------------------------------
// 1) fused projection GEMM + activations
//    grid (128, 16), block 256 (8 waves). wave -> row tile, block.y -> k tile.
//    7 WMMA accumulators per wave: chunks {A,omega,phi,alpha,g,beta} + phase.
// ---------------------------------------------------------------------------
__global__ __launch_bounds__(256) void k_proj(
    const float* __restrict__ x, const _Float16* __restrict__ Wcat,
    const float* __restrict__ b_proj, const float* __restrict__ b_phase,
    float* __restrict__ alpha_o, float* __restrict__ dr_o, float* __restrict__ di_o,
    float* __restrict__ ang_o, float* __restrict__ beta_o, float* __restrict__ g_o,
    float* __restrict__ pq_o)
{
    const int wave = threadIdx.x >> 5, lane = threadIdx.x & 31;
    const int g = lane >> 4, m = lane & 15;
    const int rowtile = blockIdx.x * 8 + wave;
    const int kcol = blockIdx.y * 16 + m;                   // oscillator index (N dim)
    const int arow = rowtile * 16 + m;                      // token row for A frag

    v8f acc[7];
#pragma unroll
    for (int c = 0; c < 7; ++c)
#pragma unroll
        for (int i = 0; i < 8; ++i) acc[c][i] = 0.f;

    const float* xrow = x + (size_t)arow * DM;
#pragma unroll 2
    for (int d = 0; d < DM; d += 32) {
        __builtin_prefetch(xrow + d + 256, 0, 0);           // global_prefetch_b8
        v16h a = load_a_f32(xrow + d, g);
#pragma unroll
        for (int c = 0; c < 7; ++c) {
            int wrow = (c < 6) ? (c * KOSC + kcol) : (1536 + kcol);
            v16h b = load_b_f16(Wcat + (size_t)wrow * DM + d, g);
            acc[c] = __builtin_amdgcn_wmma_f32_16x16x32_f16(
                false, a, false, b, (short)0, acc[c], false, false);
        }
    }

    // epilogue: bias + activations + drive, per C element (row i+8g, col kcol)
#pragma unroll
    for (int i = 0; i < 8; ++i) {
        const int r = rowtile * 16 + i + 8 * g;
        const int k = kcol;
        float a_raw  = acc[0][i] + b_proj[k];
        float w_raw  = acc[1][i] + b_proj[KOSC + k];
        float p_raw  = acc[2][i] + b_proj[2 * KOSC + k];
        float al_raw = acc[3][i] + b_proj[3 * KOSC + k];
        float g_raw  = acc[4][i] + b_proj[4 * KOSC + k];
        float be_raw = acc[5][i] + b_proj[5 * KOSC + k];
        float pq     = acc[6][i] + b_phase[k];

        float Aamp  = 3.0f * sigmoidf_(a_raw);
        float omega = softplusf_(w_raw);
        float alpha = sigmoidf_(al_raw);
        float gg    = sigmoidf_(g_raw);
        float beta  = sigmoidf_(be_raw);

        int n = r & (NSEQ - 1);
        float pos = log1pf((float)n);
        float angle = fmaf(omega, pos, p_raw);
        float s, cA;
        __sincosf(angle, &s, &cA);
        float oma = 1.f - alpha;

        size_t idx = (size_t)r * KOSC + k;
        alpha_o[idx] = alpha;
        dr_o[idx]    = oma * Aamp * cA;
        di_o[idx]    = oma * Aamp * s;
        ang_o[idx]   = angle;
        beta_o[idx]  = beta;
        g_o[idx]     = gg;
        pq_o[idx]    = pq;
    }
}

// ---------------------------------------------------------------------------
// 2a) per-chunk composition of the linear recurrence (coalesced: lane = k)
// ---------------------------------------------------------------------------
__global__ __launch_bounds__(256) void k_scan_chunks(
    const float* __restrict__ alpha, const float* __restrict__ dr,
    const float* __restrict__ di,
    float* __restrict__ ca_o, float* __restrict__ cbr_o, float* __restrict__ cbi_o)
{
    const int k = threadIdx.x, b = blockIdx.x, ch = blockIdx.y;
    float A = 1.f, br = 0.f, bi = 0.f;
    size_t base = ((size_t)(b * NSEQ + ch * CHLEN)) * KOSC + k;
    for (int j = 0; j < CHLEN; ++j) {
        size_t idx = base + (size_t)j * KOSC;
        float a = alpha[idx];
        A *= a;
        br = fmaf(a, br, dr[idx]);
        bi = fmaf(a, bi, di[idx]);
    }
    size_t c = ((size_t)b * NCHUNK + ch) * KOSC + k;
    ca_o[c] = A; cbr_o[c] = br; cbi_o[c] = bi;
}

// 2b) exclusive prefix over the 32 chunk states per (b, k)
__global__ __launch_bounds__(256) void k_scan_prefix(
    const float* __restrict__ ca, const float* __restrict__ cbr,
    const float* __restrict__ cbi,
    float* __restrict__ pr_o, float* __restrict__ pi_o)
{
    const int k = threadIdx.x, b = blockIdx.x;
    float pr = 0.f, pi = 0.f;
    for (int ch = 0; ch < NCHUNK; ++ch) {
        size_t c = ((size_t)b * NCHUNK + ch) * KOSC + k;
        pr_o[c] = pr; pi_o[c] = pi;
        float A = ca[c];
        pr = fmaf(A, pr, cbr[c]);
        pi = fmaf(A, pi, cbi[c]);
    }
}

// 2c) apply pass, fused: recurrence replay + erase + norm + demodulate +
//     phase gate + SCPM neighbor cross-products -> rho (f16, padded to 1024)
__global__ __launch_bounds__(256) void k_apply(
    const float* __restrict__ alpha, const float* __restrict__ dr,
    const float* __restrict__ di, const float* __restrict__ ang,
    const float* __restrict__ beta, const float* __restrict__ gbuf,
    const float* __restrict__ pq, const float* __restrict__ pr_in,
    const float* __restrict__ pi_in, const float* __restrict__ lambda_p,
    _Float16* __restrict__ rho)
{
    __shared__ float s_re[KOSC], s_im[KOSC], s_g[KOSC];
    const int k = threadIdx.x, b = blockIdx.x, ch = blockIdx.y;
    const float lam = lambda_p[0];
    size_t c = ((size_t)b * NCHUNK + ch) * KOSC + k;
    float pr = pr_in[c], pi = pi_in[c];

    for (int j = 0; j < CHLEN; ++j) {
        const int n = ch * CHLEN + j;
        const size_t r = (size_t)b * NSEQ + n;
        const size_t idx = r * KOSC + k;

        float a  = alpha[idx];
        float rr = fmaf(a, pr, dr[idx]);
        float ri = fmaf(a, pi, di[idx]);
        pr = rr; pi = ri;                         // un-erased state carries forward

        float s, ca_;
        __sincosf(ang[idx], &s, &ca_);
        float bet = beta[idx];
        float readout = rr * ca_ + ri * s;
        float er = rr - bet * readout * ca_;
        float ei = ri - bet * readout * s;

        float mod = sqrtf(fmaf(er, er, fmaf(ei, ei, 1e-8f)));
        float inv = 1.f / fmaxf(mod, 1.f);
        er *= inv; ei *= inv;

        float re =  er * ca_ + ei * s;
        float im = -er * s   + ei * ca_;

        float rn = sqrtf(fmaf(re, re, fmaf(im, im, 1e-8f)));
        float sq, cq;
        __sincosf(pq[idx], &sq, &cq);
        float align = (re * cq + im * sq) / rn;
        float gate = sigmoidf_(lam * align);
        re *= gate; im *= gate;

        float gg = gbuf[idx];
        s_re[k] = re; s_im[k] = im; s_g[k] = gg;
        __syncthreads();

        _Float16* row = rho + r * (size_t)DM;
        row[k]        = (_Float16)(gg * re);
        row[KOSC + k] = (_Float16)(gg * im);
        if (k < KOSC - 1) {
            float re1 = s_re[k + 1], im1 = s_im[k + 1];
            float gc = 0.5f * (gg + s_g[k + 1]);
            float xre = re * re1 - im * im1;
            float xim = re * im1 + im * re1;
            row[512 + k] = (_Float16)(gc * xre);
            row[767 + k] = (_Float16)(gc * xim);
        } else {
            row[1022] = (_Float16)0.f;            // pad columns stay zero
            row[1023] = (_Float16)0.f;
        }
        __syncthreads();
    }
}

// ---------------------------------------------------------------------------
// 3) output GEMM: out = res_scale * rho @ Wres^T   (f16 WMMA, f32 out)
//    grid (128, 64), block 256 (8 waves), wave -> row tile, block.y -> col tile
// ---------------------------------------------------------------------------
__global__ __launch_bounds__(256) void k_out(
    const _Float16* __restrict__ rho, const _Float16* __restrict__ Wres,
    const float* __restrict__ res_scale_p, float* __restrict__ out)
{
    const int wave = threadIdx.x >> 5, lane = threadIdx.x & 31;
    const int g = lane >> 4, m = lane & 15;
    const int rowtile = blockIdx.x * 8 + wave;
    const int col = blockIdx.y * 16 + m;

    v8f acc;
#pragma unroll
    for (int i = 0; i < 8; ++i) acc[i] = 0.f;

    const _Float16* arow = rho  + (size_t)(rowtile * 16 + m) * DM;
    const _Float16* brow = Wres + (size_t)col * DM;
#pragma unroll 4
    for (int kk = 0; kk < DM; kk += 32) {
        __builtin_prefetch(arow + kk + 256, 0, 0);
        v16h a = load_a_f16(arow + kk, g);
        v16h b = load_b_f16(brow + kk, g);
        acc = __builtin_amdgcn_wmma_f32_16x16x32_f16(
            false, a, false, b, (short)0, acc, false, false);
    }

    const float rs = res_scale_p[0];
#pragma unroll
    for (int i = 0; i < 8; ++i) {
        int r = rowtile * 16 + i + 8 * g;
        out[(size_t)r * DM + col] = rs * acc[i];
    }
}

// ---------------------------------------------------------------------------
// launch
// ---------------------------------------------------------------------------
extern "C" void kernel_launch(void* const* d_in, const int* in_sizes, int n_in,
                              void* d_out, int out_size, void* d_ws, size_t ws_size,
                              hipStream_t stream) {
    const float* x       = (const float*)d_in[0];
    const float* W_proj  = (const float*)d_in[1];
    const float* b_proj  = (const float*)d_in[2];
    const float* W_res   = (const float*)d_in[3];
    const float* W_phase = (const float*)d_in[4];
    const float* b_phase = (const float*)d_in[5];
    const float* lam     = (const float*)d_in[6];
    const float* rscale  = (const float*)d_in[7];
    float* out = (float*)d_out;

    char* ws = (char*)d_ws;
    size_t off = 0;
    auto take = [&](size_t bytes) { char* p = ws + off; off += (bytes + 255) & ~(size_t)255; return p; };

    _Float16* Wcat  = (_Float16*)take(1792ull * DM * 2);
    _Float16* Wresh = (_Float16*)take((size_t)DM * DM * 2);
    float* alpha = (float*)take((size_t)MROW * KOSC * 4);
    float* drv_r = (float*)take((size_t)MROW * KOSC * 4);
    float* drv_i = (float*)take((size_t)MROW * KOSC * 4);
    float* angb  = (float*)take((size_t)MROW * KOSC * 4);
    float* betab = (float*)take((size_t)MROW * KOSC * 4);
    float* gb    = (float*)take((size_t)MROW * KOSC * 4);
    float* pqb   = (float*)take((size_t)MROW * KOSC * 4);
    float* ca    = (float*)take((size_t)NB * NCHUNK * KOSC * 4);
    float* cbr   = (float*)take((size_t)NB * NCHUNK * KOSC * 4);
    float* cbi   = (float*)take((size_t)NB * NCHUNK * KOSC * 4);
    float* prb   = (float*)take((size_t)NB * NCHUNK * KOSC * 4);
    float* pib   = (float*)take((size_t)NB * NCHUNK * KOSC * 4);
    _Float16* rho = (_Float16*)take((size_t)MROW * DM * 2);

    // 0) pack weights to f16
    k_pack_wcat<<<dim3((1792 * DM + 255) / 256), dim3(256), 0, stream>>>(W_proj, W_phase, Wcat);
    k_pack_wres<<<dim3((DM * DM + 255) / 256), dim3(256), 0, stream>>>(W_res, Wresh);

    // 1) fused projection (6 chunks + phase query) + activations
    k_proj<<<dim3(MROW / 128, KOSC / 16), dim3(256), 0, stream>>>(
        x, Wcat, b_proj, b_phase, alpha, drv_r, drv_i, angb, betab, gb, pqb);

    // 2) chunked parallel scan + fused post-processing -> rho (f16)
    k_scan_chunks<<<dim3(NB, NCHUNK), dim3(KOSC), 0, stream>>>(alpha, drv_r, drv_i, ca, cbr, cbi);
    k_scan_prefix<<<dim3(NB), dim3(KOSC), 0, stream>>>(ca, cbr, cbi, prb, pib);
    k_apply<<<dim3(NB, NCHUNK), dim3(KOSC), 0, stream>>>(
        alpha, drv_r, drv_i, angb, betab, gb, pqb, prb, pib, lam, rho);

    // 3) output GEMM
    k_out<<<dim3(MROW / 128, DM / 16), dim3(256), 0, stream>>>(rho, Wresh, rscale, out);
}
